// TitanLayer_70205535420756
// MI455X (gfx1250) — matrix operand order repaired and checked by reading, not compiled, not run on hardware
//
#include <hip/hip_runtime.h>
#include <hip/hip_bf16.h>
#include <hip/hip_fp16.h>

#define B_     4
#define L_     4096
#define D_     1024
#define H_     16
#define HD_    64
#define CHUNK_ 64
#define NC_    (L_/CHUNK_)
#define M_TOT  (B_*L_)

typedef __attribute__((ext_vector_type(16))) _Float16 v16h;
typedef __attribute__((ext_vector_type(8)))  _Float16 v8h;
typedef __attribute__((ext_vector_type(8)))  float    v8f;
typedef __attribute__((ext_vector_type(4)))  int      v4i;

#define AS1 __attribute__((address_space(1)))
#define AS3 __attribute__((address_space(3)))

#if __has_builtin(__builtin_amdgcn_global_load_async_to_lds_b128)
#define HAVE_ASYNC 1
#else
#define HAVE_ASYNC 0
#endif

__device__ __forceinline__ void async_copy16(const _Float16* src, _Float16* dst) {
#if HAVE_ASYNC
    // 16 f16 = 2x b128 async global->LDS copies (ASYNCcnt path)
    __builtin_amdgcn_global_load_async_to_lds_b128((AS1 v4i*)src,       (AS3 v4i*)dst,       0, 0);
    __builtin_amdgcn_global_load_async_to_lds_b128((AS1 v4i*)(src + 8), (AS3 v4i*)(dst + 8), 0, 0);
#else
    *(v8h*)dst       = *(const v8h*)src;
    *(v8h*)(dst + 8) = *(const v8h*)(src + 8);
#endif
}

__device__ __forceinline__ void async_wait0() {
#if HAVE_ASYNC
#if __has_builtin(__builtin_amdgcn_s_wait_asynccnt)
    __builtin_amdgcn_s_wait_asynccnt(0);
#else
    asm volatile("s_wait_asynccnt 0x0" ::: "memory");
#endif
#endif
}

__device__ __forceinline__ float sigmoidf_(float x) { return 1.0f / (1.0f + __expf(-x)); }
__device__ __forceinline__ v8f   zero8f_() { v8f z = {}; return z; }

// ---- WMMA fragment helpers (CDNA5 wave32 layouts, 16x16x32 f16) ----------
// A fragment: 16x32 (MxK). Lane m=lane&15, half=lane>>4.
// VGPR0..3 hold K = half*8 + 0..7 ; VGPR4..7 hold K = 16 + half*8 + 0..7.
__device__ __forceinline__ v16h load_a_frag(const _Float16* base, int ld, int lane) {
    const int m = lane & 15, half = lane >> 4;
    const _Float16* p = base + m * ld + half * 8;
    v16h a;
#pragma unroll
    for (int i = 0; i < 8; i++) a[i] = p[i];
#pragma unroll
    for (int i = 0; i < 8; i++) a[8 + i] = p[16 + i];
    return a;
}
// B fragment: 32x16 (KxN), read from a transposed image BT[n][k] (row-major in n).
// Lane n=lane&15 holds K = (lane>>4)*8 + 0..7 and 16 + (lane>>4)*8 + 0..7.
__device__ __forceinline__ v16h load_bt_frag(const _Float16* base, int ld, int lane) {
    const int n = lane & 15, kh = lane >> 4;
    const _Float16* p = base + n * ld + kh * 8;
    v16h b;
#pragma unroll
    for (int i = 0; i < 8; i++) { b[i] = p[i]; b[8 + i] = p[16 + i]; }
    return b;
}

// ---- f32 -> f16 weight conversion (once per weight matrix) ----------------
__global__ void cvt_w_kernel(const float* __restrict__ W, _Float16* __restrict__ Wh, int n) {
    int i = blockIdx.x * blockDim.x + threadIdx.x;
    if (i < n) Wh[i] = (_Float16)W[i];
}

// ---- depthwise conv(k=4,pad=2)[:L] + bias + SiLU -> f16 -------------------
__global__ void conv_silu_kernel(const float* __restrict__ x, const float* __restrict__ cw,
                                 const float* __restrict__ cb, _Float16* __restrict__ XH) {
    size_t idx = (size_t)blockIdx.x * blockDim.x + threadIdx.x;
    if (idx >= (size_t)M_TOT * D_) return;
    const int d = (int)(idx % D_);
    const size_t bl = idx / D_;
    const int l = (int)(bl % L_);
    const int b = (int)(bl / L_);
    float acc = cb[d];
#pragma unroll
    for (int t = 0; t < 4; t++) {
        const int ll = l + t - 2;
        if (ll >= 0 && ll < L_)
            acc += x[((size_t)b * L_ + ll) * D_ + d] * cw[d * 4 + t];
    }
    XH[idx] = (_Float16)(acc * sigmoidf_(acc));
}

// ---- tiled WMMA GEMM: C[M,N] = A[M,K] * Wh[N,K]^T, both f16 ---------------
// 256 threads = 8 waves; 128x128 block tile; K-step 32; wave tile 32x64.
// Double-buffered LDS fed by async global->LDS b128 copies.
template <bool HALF_OUT>
__global__ __launch_bounds__(256) void gemm_xwT(const _Float16* __restrict__ A,
                                                const _Float16* __restrict__ Wh,
                                                _Float16* __restrict__ Ch,
                                                float* __restrict__ Cf,
                                                int M, int N, int K) {
    __shared__ _Float16 As[2][128][48];  // [buf][m][k]
    __shared__ _Float16 Bs[2][128][48];  // [buf][n][k]
    const int tid = threadIdx.x, lane = tid & 31, wave = tid >> 5;
    const int bm = blockIdx.y * 128, bn = blockIdx.x * 128;
    const int waveM = (wave & 3) * 32, waveN = (wave >> 2) * 64;

    v8f acc[2][4];
#pragma unroll
    for (int i = 0; i < 2; i++)
#pragma unroll
        for (int j = 0; j < 4; j++) acc[i][j] = zero8f_();

    const int lr = tid >> 1;        // 0..127
    const int lc = (tid & 1) * 16;  // 0 / 16
    const _Float16* Ab = A  + (size_t)(bm + lr) * K + lc;
    const _Float16* Bb = Wh + (size_t)(bn + lr) * K + lc;

    const int nk = K / 32;
    // prologue: stage K-tile 0 into buffer 0
    async_copy16(Ab, &As[0][lr][lc]);
    async_copy16(Bb, &Bs[0][lr][lc]);

    for (int it = 0; it < nk; ++it) {
        const int cur = it & 1;
        async_wait0();
        __syncthreads();  // tile `cur` resident; everyone done with buffer `cur^1`
        if (it + 1 < nk) {
            async_copy16(Ab + (it + 1) * 32, &As[cur ^ 1][lr][lc]);
            async_copy16(Bb + (it + 1) * 32, &Bs[cur ^ 1][lr][lc]);
        }
        v16h aF[2];
#pragma unroll
        for (int mi = 0; mi < 2; mi++) aF[mi] = load_a_frag(&As[cur][waveM + 16 * mi][0], 48, lane);
#pragma unroll
        for (int ni = 0; ni < 4; ni++) {
            v16h bF = load_bt_frag(&Bs[cur][waveN + 16 * ni][0], 48, lane);
#pragma unroll
            for (int mi = 0; mi < 2; mi++)
                acc[mi][ni] = __builtin_amdgcn_wmma_f32_16x16x32_f16(
                    false, aF[mi], false, bF, (short)0, acc[mi][ni], false, false);
        }
    }

    const int nref = lane & 15, mh = (lane >> 4) * 8;
#pragma unroll
    for (int mi = 0; mi < 2; mi++)
#pragma unroll
        for (int ni = 0; ni < 4; ni++)
#pragma unroll
            for (int r = 0; r < 8; r++) {
                const int row = bm + waveM + 16 * mi + mh + r;
                const int col = bn + waveN + 16 * ni + nref;
                if (HALF_OUT) Ch[(size_t)row * N + col] = (_Float16)acc[mi][ni][r];
                else          Cf[(size_t)row * N + col] = acc[mi][ni][r];
            }
}

// ---- gating + surprise ----------------------------------------------------
__global__ __launch_bounds__(64) void gate_surprise_kernel(
    const _Float16* __restrict__ KH, _Float16* __restrict__ VH,
    const _Float16* __restrict__ GH, const float* __restrict__ Wsp,
    const float* __restrict__ bsp, _Float16* __restrict__ KS, _Float16* __restrict__ VS) {
    __shared__ float red[64];
    const int t = threadIdx.x;  // d in [0,64)
    const size_t blh = blockIdx.x;
    const int h = (int)(blh % H_);
    const size_t bl = blh / H_;
    const size_t off = bl * D_ + (size_t)h * HD_ + t;
    const float k = (float)KH[off];
    const float v = (float)VH[off] * sigmoidf_((float)GH[off]);
    red[t] = k * Wsp[t] + v * Wsp[HD_ + t];
    __syncthreads();
    for (int s = 32; s > 0; s >>= 1) {
        if (t < s) red[t] += red[t + s];
        __syncthreads();
    }
    const float sur = sigmoidf_((red[0] + bsp[0]) * 10.0f);
    VH[off] = (_Float16)v;
    KS[off] = (_Float16)(k * sur);
    VS[off] = (_Float16)(v * sur);
}

// ---- chunked linear-attention scan ---------------------------------------
// One block (8 waves) per (b,h). State S (64x64 f32) lives in VGPRs across all
// 64 chunks; per chunk: S = f*S + V^T K (WMMA), O = Q S (WMMA via f16 S^T in LDS).
__global__ __launch_bounds__(256) void chunk_scan_kernel(
    const _Float16* __restrict__ Qg, const _Float16* __restrict__ Kg,
    const _Float16* __restrict__ Vg, const float* __restrict__ decay,
    _Float16* __restrict__ Og) {
    __shared__ _Float16 Qs[64][72];  // [l][d]
    __shared__ _Float16 KT[64][72];  // [e][l]
    __shared__ _Float16 VT[64][72];  // [d][l]
    __shared__ _Float16 ST[64][72];  // [e][d]
    __shared__ float red[256];
    __shared__ float sFactor;

    const int tid = threadIdx.x, lane = tid & 31, wave = tid >> 5;
    const int bh = blockIdx.x, b = bh / H_, h = bh % H_;

    // factor = mean(sigmoid(decay over H*HD=1024 vals)) ^ CHUNK
    {
        float p = 0.f;
#pragma unroll
        for (int i = 0; i < 4; i++) p += sigmoidf_(decay[tid * 4 + i]);
        red[tid] = p;
        __syncthreads();
        for (int s = 128; s > 0; s >>= 1) {
            if (tid < s) red[tid] += red[tid + s];
            __syncthreads();
        }
        if (tid == 0) sFactor = __powf(red[0] * (1.0f / 1024.0f), (float)CHUNK_);
        __syncthreads();
    }
    const float factor = sFactor;

    const int mT = (wave & 3) * 16;   // row-tile origin
    const int nT = (wave >> 2) * 32;  // col origin (2 tiles per wave)
    v8f S[2];
    S[0] = zero8f_(); S[1] = zero8f_();

    const int lrow = tid >> 2;        // 0..63
    const int lcol = (tid & 3) * 16;  // 0,16,32,48
    const int nref = lane & 15, mh = (lane >> 4) * 8;

    for (int c = 0; c < NC_; c++) {
        const size_t gb = ((size_t)b * L_ + (size_t)c * CHUNK_ + lrow) * D_ + (size_t)h * HD_ + lcol;
        // Q tile is a straight row-major copy -> async b128 path
        async_copy16(Qg + gb, &Qs[lrow][lcol]);
        // K/V tiles are transposed into LDS: vector global loads, scalar ds stores
        {
            v8h k0 = *(const v8h*)(Kg + gb), k1 = *(const v8h*)(Kg + gb + 8);
            v8h v0 = *(const v8h*)(Vg + gb), v1 = *(const v8h*)(Vg + gb + 8);
#pragma unroll
            for (int i = 0; i < 8; i++) {
                KT[lcol + i][lrow]     = k0[i];
                KT[lcol + 8 + i][lrow] = k1[i];
                VT[lcol + i][lrow]     = v0[i];
                VT[lcol + 8 + i][lrow] = v1[i];
            }
        }
        if (c + 1 < NC_) {  // hide the serial chunk dependence
            __builtin_prefetch(Qg + gb + (size_t)CHUNK_ * D_, 0, 1);
            __builtin_prefetch(Kg + gb + (size_t)CHUNK_ * D_, 0, 1);
            __builtin_prefetch(Vg + gb + (size_t)CHUNK_ * D_, 0, 1);
        }
        async_wait0();
        __syncthreads();

        // S = factor*S + V^T K
#pragma unroll
        for (int ni = 0; ni < 2; ni++)
#pragma unroll
            for (int r = 0; r < 8; r++) S[ni][r] *= factor;
#pragma unroll
        for (int l0 = 0; l0 < 64; l0 += 32) {
            v16h aV = load_a_frag(&VT[mT][l0], 72, lane);
#pragma unroll
            for (int ni = 0; ni < 2; ni++) {
                v16h bK = load_bt_frag(&KT[nT + 16 * ni][l0], 72, lane);
                S[ni] = __builtin_amdgcn_wmma_f32_16x16x32_f16(
                    false, aV, false, bK, (short)0, S[ni], false, false);
            }
        }
        // publish S^T as f16 for the O gemm
#pragma unroll
        for (int ni = 0; ni < 2; ni++)
#pragma unroll
            for (int r = 0; r < 8; r++)
                ST[nT + 16 * ni + nref][mT + mh + r] = (_Float16)S[ni][r];
        __syncthreads();

        // O = Q @ S   (queries read post-update state)
        v8f O[2];
        O[0] = zero8f_(); O[1] = zero8f_();
#pragma unroll
        for (int d0 = 0; d0 < 64; d0 += 32) {
            v16h aQ = load_a_frag(&Qs[mT][d0], 72, lane);
#pragma unroll
            for (int ni = 0; ni < 2; ni++) {
                v16h bS = load_bt_frag(&ST[nT + 16 * ni][d0], 72, lane);
                O[ni] = __builtin_amdgcn_wmma_f32_16x16x32_f16(
                    false, aQ, false, bS, (short)0, O[ni], false, false);
            }
        }
#pragma unroll
        for (int ni = 0; ni < 2; ni++)
#pragma unroll
            for (int r = 0; r < 8; r++) {
                const int l = c * CHUNK_ + mT + mh + r;
                const int e = nT + 16 * ni + nref;
                Og[((size_t)b * L_ + l) * D_ + (size_t)h * HD_ + e] = (_Float16)O[ni][r];
            }
        __syncthreads();
    }
}

// ---- alpha blend ----------------------------------------------------------
__global__ void blend_kernel(const _Float16* __restrict__ OFp, const _Float16* __restrict__ OSp,
                             const float* __restrict__ alpha, _Float16* __restrict__ OUT) {
    size_t i = (size_t)blockIdx.x * blockDim.x + threadIdx.x;
    if (i >= (size_t)M_TOT * D_) return;
    const float a = sigmoidf_(alpha[0]);
    OUT[i] = (_Float16)(a * (float)OFp[i] + (1.0f - a) * (float)OSp[i]);
}

extern "C" void kernel_launch(void* const* d_in, const int* in_sizes, int n_in,
                              void* d_out, int out_size, void* d_ws, size_t ws_size,
                              hipStream_t stream) {
    (void)in_sizes; (void)n_in; (void)out_size; (void)ws_size;
    const float* x  = (const float*)d_in[0];
    const float* Wq = (const float*)d_in[1];
    const float* Wk = (const float*)d_in[2];
    const float* Wv = (const float*)d_in[3];
    const float* Wo = (const float*)d_in[4];
    const float* Wg = (const float*)d_in[5];
    const float* cw = (const float*)d_in[6];
    const float* cb = (const float*)d_in[7];
    const float* Ws = (const float*)d_in[8];
    const float* bs = (const float*)d_in[9];
    const float* fd = (const float*)d_in[10];
    const float* sd = (const float*)d_in[11];
    const float* al = (const float*)d_in[12];

    const size_t E    = (size_t)M_TOT * D_;       // 16,777,216 elems
    const size_t REG  = E * sizeof(_Float16);     // 32 MiB per region
    const size_t WREG = (size_t)D_ * D_ * sizeof(_Float16);  // 2 MiB per weight
    char* w = (char*)d_ws;
    _Float16* XH = (_Float16*)(w + 0 * REG);
    _Float16* QH = (_Float16*)(w + 1 * REG);
    _Float16* KH = (_Float16*)(w + 2 * REG);
    _Float16* VH = (_Float16*)(w + 3 * REG);
    _Float16* GH = (_Float16*)(w + 4 * REG);
    _Float16* KS = (_Float16*)(w + 5 * REG);
    _Float16* VS = (_Float16*)(w + 6 * REG);
    _Float16* WhQ = (_Float16*)(w + 7 * REG + 0 * WREG);
    _Float16* WhK = (_Float16*)(w + 7 * REG + 1 * WREG);
    _Float16* WhV = (_Float16*)(w + 7 * REG + 2 * WREG);
    _Float16* WhG = (_Float16*)(w + 7 * REG + 3 * WREG);
    _Float16* WhO = (_Float16*)(w + 7 * REG + 4 * WREG);
    _Float16* OFAST = XH;   // XH dead after projections
    _Float16* OSLOW = GH;   // GH dead after gating
    _Float16* OF    = KH;   // KH dead after fast scan

    const int ew_blocks = (int)((E + 255) / 256);
    const int wcnt = D_ * D_;
    const int w_blocks = (wcnt + 255) / 256;
    conv_silu_kernel<<<ew_blocks, 256, 0, stream>>>(x, cw, cb, XH);
    cvt_w_kernel<<<w_blocks, 256, 0, stream>>>(Wq, WhQ, wcnt);
    cvt_w_kernel<<<w_blocks, 256, 0, stream>>>(Wk, WhK, wcnt);
    cvt_w_kernel<<<w_blocks, 256, 0, stream>>>(Wv, WhV, wcnt);
    cvt_w_kernel<<<w_blocks, 256, 0, stream>>>(Wg, WhG, wcnt);
    cvt_w_kernel<<<w_blocks, 256, 0, stream>>>(Wo, WhO, wcnt);

    dim3 ggrid(D_ / 128, M_TOT / 128);  // (8, 128)
    gemm_xwT<true><<<ggrid, 256, 0, stream>>>(XH, WhQ, QH, nullptr, M_TOT, D_, D_);
    gemm_xwT<true><<<ggrid, 256, 0, stream>>>(XH, WhK, KH, nullptr, M_TOT, D_, D_);
    gemm_xwT<true><<<ggrid, 256, 0, stream>>>(XH, WhV, VH, nullptr, M_TOT, D_, D_);
    gemm_xwT<true><<<ggrid, 256, 0, stream>>>(XH, WhG, GH, nullptr, M_TOT, D_, D_);

    gate_surprise_kernel<<<(size_t)B_ * L_ * H_, 64, 0, stream>>>(KH, VH, GH, Ws, bs, KS, VS);

    chunk_scan_kernel<<<B_ * H_, 256, 0, stream>>>(QH, KH, VH, fd, OFAST);
    chunk_scan_kernel<<<B_ * H_, 256, 0, stream>>>(QH, KS, VS, sd, OSLOW);

    blend_kernel<<<ew_blocks, 256, 0, stream>>>(OFAST, OSLOW, al, OF);

    gemm_xwT<false><<<ggrid, 256, 0, stream>>>(OF, WhO, nullptr, (float*)d_out, M_TOT, D_, D_);
}
